// SparseAttentionModule_55645596287221
// MI455X (gfx1250) — matrix operand order repaired and checked by reading, not compiled, not run on hardware
//
#include <hip/hip_runtime.h>
#include <hip/hip_bf16.h>
#include <stdint.h>

typedef __attribute__((ext_vector_type(16))) _Float16 v16h;
typedef __attribute__((ext_vector_type(8)))  float    v8f;

#define INDIM   128
#define OUTDIM  128
#define EDIM    32
#define NHEADS  8
#define HDIM    16
#define SCALE_Q 0.25f          // HEAD_DIM^-0.5, folded into Wq/bq
#define NTILES  24             // 384 output cols (Q|K|V) / 16
#define KTILES  4              // 128 / 32
#define NTGROUP 8              // col tiles swept per wave (A-fragment reuse)

// ---------------------------------------------------------------- zero output
__global__ void zero_f32(float* __restrict__ p, int n) {
  int i = blockIdx.x * blockDim.x + threadIdx.x;
  int stride = gridDim.x * blockDim.x;
  for (; i < n; i += stride) p[i] = 0.0f;
}

// --------------------------------------------------- pack weights / reductions
// WhB layout: [nt][ks][lane][16] f16, pre-swizzled into the CDNA5 16-bit
// B-matrix (32x16) lane layout: lane<16 -> N=lane, K = ks*32 + {0..7,16..23};
//                               lane>=16 -> N=lane-16, K = ks*32 + {8..15,24..31}.
__global__ void prep_kernel(const float* __restrict__ Wq, const float* __restrict__ bq,
                            const float* __restrict__ Wk, const float* __restrict__ bk,
                            const float* __restrict__ Wv, const float* __restrict__ bv,
                            const float* __restrict__ We, const float* __restrict__ be,
                            _Float16* __restrict__ WhB, float* __restrict__ biasAll,
                            float* __restrict__ WerT, float* __restrict__ ber) {
  // pack Q|K|V weights (SCALE folded into Q)
  for (int t = threadIdx.x; t < NTILES * KTILES * 32 * 16; t += blockDim.x) {
    int i    = t & 15;
    int lane = (t >> 4) & 31;
    int ks   = (t >> 9) & 3;
    int nt   = t >> 11;
    int n = nt * 16 + (lane & 15);
    int k = ks * 32 + ((lane & 16) ? 8 : 0) + (i & 7) + ((i & 8) ? 16 : 0);
    float w;
    if (n < 128)      w = Wq[k * 128 + n] * SCALE_Q;
    else if (n < 256) w = Wk[k * 128 + (n - 128)];
    else              w = Wv[k * 128 + (n - 256)];
    WhB[t] = (_Float16)w;
  }
  for (int t = threadIdx.x; t < 3 * OUTDIM; t += blockDim.x) {
    biasAll[t] = (t < 128) ? bq[t] * SCALE_Q
               : (t < 256) ? bk[t - 128] : bv[t - 256];
  }
  // reduced edge weights: WerT[h][c] = sum_d We[c][h*16+d]  (8 x 32, row-major)
  for (int t = threadIdx.x; t < NHEADS * EDIM; t += blockDim.x) {
    int h = t / EDIM, c = t % EDIM;
    float s = 0.0f;
    for (int d = 0; d < HDIM; ++d) s += We[c * OUTDIM + h * HDIM + d];
    WerT[t] = s;
  }
  for (int t = threadIdx.x; t < NHEADS; t += blockDim.x) {
    float s = 0.0f;
    for (int d = 0; d < HDIM; ++d) s += be[t * HDIM + d];
    ber[t] = s;
  }
}

// ------------------------------------------------------- QKV projection (WMMA)
// Each wave owns one 16-row A strip (converted f32->f16 ONCE into 32 VGPRs)
// and sweeps NTGROUP=8 column tiles with it: 32 WMMAs per A conversion.
__global__ void qkv_gemm_kernel(const float* __restrict__ X,
                                const _Float16* __restrict__ WhB,
                                const float* __restrict__ biasAll,
                                float* __restrict__ Q, float* __restrict__ K,
                                float* __restrict__ V, int ntasks) {
  int tid  = blockIdx.x * 8 + (threadIdx.x >> 5);
  int lane = threadIdx.x & 31;
  if (tid >= ntasks) return;                 // wave-uniform
  const int NGRP = NTILES / NTGROUP;         // 3 col-tile groups
  int mt = tid / NGRP;
  int g  = tid % NGRP;

  // A-matrix 16x32 f16 lane layout: lane<16 holds row=lane, K {0..7,16..23};
  //                                 lane>=16 holds row=lane-16, K {8..15,24..31}
  int row  = mt * 16 + (lane & 15);
  int koff = (lane & 16) ? 8 : 0;
  const float* xrow = X + (size_t)row * INDIM;

  v16h afrag[KTILES];
#pragma unroll
  for (int ks = 0; ks < KTILES; ++ks) {
    const float* ap = xrow + ks * 32 + koff;
    float4 a0 = *(const float4*)(ap + 0);
    float4 a1 = *(const float4*)(ap + 4);
    float4 a2 = *(const float4*)(ap + 16);
    float4 a3 = *(const float4*)(ap + 20);
    v16h a;
    a[0]=(_Float16)a0.x;  a[1]=(_Float16)a0.y;  a[2]=(_Float16)a0.z;  a[3]=(_Float16)a0.w;
    a[4]=(_Float16)a1.x;  a[5]=(_Float16)a1.y;  a[6]=(_Float16)a1.z;  a[7]=(_Float16)a1.w;
    a[8]=(_Float16)a2.x;  a[9]=(_Float16)a2.y;  a[10]=(_Float16)a2.z; a[11]=(_Float16)a2.w;
    a[12]=(_Float16)a3.x; a[13]=(_Float16)a3.y; a[14]=(_Float16)a3.z; a[15]=(_Float16)a3.w;
    afrag[ks] = a;
  }

  int mbase = mt * 16 + ((lane & 16) ? 8 : 0);

#pragma unroll
  for (int j = 0; j < NTGROUP; ++j) {
    int nt = g * NTGROUP + j;
    v8f acc = {};
#pragma unroll
    for (int ks = 0; ks < KTILES; ++ks) {
      v16h b = *(const v16h*)(WhB + ((size_t)(nt * KTILES + ks) * 32 + lane) * 16);
      acc = __builtin_amdgcn_wmma_f32_16x16x32_f16(
          /*neg_a=*/false, afrag[ks], /*neg_b=*/false, b,
          /*c_mod=*/(short)0, acc, /*reuse_a=*/false, /*reuse_b=*/false);
    }
    // C/D layout: VGPR r -> row (r | r+8), N = lane%16
    int col   = nt * 16 + (lane & 15);
    float bias = biasAll[col];
    float* dstp; int c;
    if (col < 128)      { dstp = Q; c = col; }
    else if (col < 256) { dstp = K; c = col - 128; }
    else                { dstp = V; c = col - 256; }
#pragma unroll
    for (int r = 0; r < 8; ++r)
      dstp[(size_t)(mbase + r) * OUTDIM + c] = acc[r] + bias;
  }
}

// ---------------------------------------------------- edge phase (1 wave/edge)
__global__ void edge_kernel(const float* __restrict__ ef,
                            const int* __restrict__ edge_list,
                            const float* __restrict__ Q, const float* __restrict__ K,
                            const float* __restrict__ V,
                            const float* __restrict__ WerT, const float* __restrict__ ber,
                            float* __restrict__ out, int n_edges) {
  int e    = blockIdx.x * 8 + (threadIdx.x >> 5);
  int lane = threadIdx.x & 31;
  if (e >= n_edges) return;                  // wave-uniform
  int src = edge_list[2 * e + 0];
  int dst = edge_list[2 * e + 1];
  int h   = lane >> 2;                       // head of this lane's 4 dims

  // Q·K per head (SCALE already folded into Q)
  float4 q = *(const float4*)(Q + (size_t)src * OUTDIM + lane * 4);
  float4 k = *(const float4*)(K + (size_t)dst * OUTDIM + lane * 4);
  float qk = q.x * k.x + q.y * k.y + q.z * k.z + q.w * k.w;
  qk += __shfl_xor(qk, 1, 32);
  qk += __shfl_xor(qk, 2, 32);

  // reduced edge bias: broadcast 32 edge features via shuffles
  float efv = ef[(size_t)e * EDIM + lane];
  float eb  = ber[h];
  const float* wrow = WerT + h * EDIM;
#pragma unroll
  for (int c = 0; c < EDIM; ++c)
    eb = fmaf(__shfl(efv, c, 32), wrow[c], eb);

  float s = qk + eb;

  // softmax over 8 heads (each head duplicated x4 across the wave)
  float m = s;
#pragma unroll
  for (int o = 16; o >= 1; o >>= 1) m = fmaxf(m, __shfl_xor(m, o, 32));
  float ex  = __expf(s - m);
  float sum = ex;
#pragma unroll
  for (int o = 16; o >= 1; o >>= 1) sum += __shfl_xor(sum, o, 32);
  float w = ex * 4.0f / sum;                 // correct x4 duplication

  float4 v = *(const float4*)(V + (size_t)src * OUTDIM + lane * 4);
  float* op = out + (size_t)dst * OUTDIM + lane * 4;
  atomicAdd(op + 0, v.x * w);
  atomicAdd(op + 1, v.y * w);
  atomicAdd(op + 2, v.z * w);
  atomicAdd(op + 3, v.w * w);
}

// ------------------------------------------------------------------- launcher
extern "C" void kernel_launch(void* const* d_in, const int* in_sizes, int n_in,
                              void* d_out, int out_size, void* d_ws, size_t ws_size,
                              hipStream_t stream) {
  const float* X  = (const float*)d_in[0];
  const float* EF = (const float*)d_in[1];
  const float* Wq = (const float*)d_in[2];
  const float* bq = (const float*)d_in[3];
  const float* Wk = (const float*)d_in[4];
  const float* bk = (const float*)d_in[5];
  const float* Wv = (const float*)d_in[6];
  const float* bv = (const float*)d_in[7];
  const float* We = (const float*)d_in[8];
  const float* be = (const float*)d_in[9];
  const int* edge_list = (const int*)d_in[10];
  float* out = (float*)d_out;

  int nodes = in_sizes[0] / INDIM;     // 50000 (multiple of 16)
  int edges = in_sizes[10] / 2;        // 800000

  // workspace layout
  float* Q  = (float*)d_ws;
  float* Km = Q  + (size_t)nodes * OUTDIM;
  float* Vm = Km + (size_t)nodes * OUTDIM;
  float* biasAll = Vm + (size_t)nodes * OUTDIM;
  float* WerT    = biasAll + 3 * OUTDIM;
  float* ber     = WerT + NHEADS * EDIM;
  uintptr_t p = (uintptr_t)(ber + NHEADS);
  p = (p + 63) & ~(uintptr_t)63;       // 64B-align for v16h loads
  _Float16* WhB = (_Float16*)p;

  zero_f32<<<2048, 256, 0, stream>>>(out, nodes * OUTDIM);
  prep_kernel<<<1, 256, 0, stream>>>(Wq, bq, Wk, bk, Wv, bv, We, be,
                                     WhB, biasAll, WerT, ber);
  int ntasks = (nodes / 16) * (NTILES / NTGROUP);   // 9375 waves
  qkv_gemm_kernel<<<(ntasks + 7) / 8, 256, 0, stream>>>(X, WhB, biasAll,
                                                        Q, Km, Vm, ntasks);
  edge_kernel<<<(edges + 7) / 8, 256, 0, stream>>>(EF, edge_list, Q, Km, Vm,
                                                   WerT, ber, out, edges);
}
